// ChannelwiseSlidingWindowDropout2D_48790828482807
// MI455X (gfx1250) — matrix operand order repaired and checked by reading, not compiled, not run on hardware
//
#include <hip/hip_runtime.h>

// ---------------------------------------------------------------------------
// ChannelwiseSlidingWindowDropout2D for MI455X (gfx1250, wave32)
//
// Phase 1 (mask_kernel): per-channel 7x7 binary dilation expressed as
//   M = T^T * Bits * T  with banded ones matrix T[k][j] = (k <= j <= k+6),
//   computed with v_wmma_f32_16x16x32_f16 (counts are exact in f16/f32).
//   Result stored as 1 byte per (c,y,x) keep-multiplier in d_ws (1 MB).
// Phase 2 (apply_kernel): bandwidth-bound streaming multiply,
//   non-temporal B128 load/store for x/out, cached mask reads (L2-resident),
//   global_prefetch_b8 read-ahead.
// ---------------------------------------------------------------------------

typedef __attribute__((ext_vector_type(16))) _Float16 v16h;
typedef __attribute__((ext_vector_type(8)))  float    v8f;
typedef __attribute__((ext_vector_type(4)))  float    v4f;

#define N_CH   256
#define HWDIM  64
#define NVALID 58          // H - WINDOW + 1
#define DROPP  0.1f
#define MASK_ELEMS (N_CH * HWDIM * HWDIM)   // 1,048,576 (2^20)

// A-matrix (16x32 f16) lane layout, ISA 7.12.2: within a lane's 16 halfs,
// halfs 0..7 -> K = {0..7 | 8..15} (by lane group), halfs 8..15 -> K + 16.
__device__ __forceinline__ int a_kmap(int i, int grp) {
  return i + 8 * grp + ((i >= 8) ? 8 : 0);
}
// B-matrix (32x16 f16): same K order as A: halfs 0..7 -> K = {0..7 | 8..15}
// (by lane group), halfs 8..15 -> K + 16.
__device__ __forceinline__ int b_kmap(int i, int grp) {
  return i + 8 * grp + ((i >= 8) ? 8 : 0);
}
// Banded dilation matrix: T[r][c] = 1 iff r <= c <= r+6 (window width 7).
__device__ __forceinline__ _Float16 tval(int r, int c) {
  return (_Float16)((r <= c && c <= r + 6) ? 1.0f : 0.0f);
}

__global__ void __launch_bounds__(128)
mask_kernel(const float* __restrict__ noise, unsigned char* __restrict__ maskb) {
  __shared__ _Float16 Bits[HWDIM * HWDIM];  // 8 KB: window-start bits, zero-padded
  __shared__ _Float16 Hm[HWDIM * HWDIM];    // 8 KB: after horizontal pass

  const int c   = blockIdx.x;
  const int tid = threadIdx.x;

  // 1) bits = (noise < p), zero-padded 58x58 -> 64x64 (f16, exact 0/1)
  const float* nz = noise + (size_t)c * (NVALID * NVALID);
  for (int idx = tid; idx < HWDIM * HWDIM; idx += 128) {
    int y = idx >> 6, x = idx & 63;
    float v = 0.0f;
    if (y < NVALID && x < NVALID) v = (nz[y * NVALID + x] < DROPP) ? 1.0f : 0.0f;
    Bits[idx] = (_Float16)v;
  }
  __syncthreads();

  const int wave = tid >> 5;     // 4 waves: wave w owns output tile-row mt = w
  const int lane = tid & 31;
  const int grp  = lane >> 4;    // lane group (0: lanes 0-15, 1: lanes 16-31)
  const int ln16 = lane & 15;
  const int mt   = wave;

  // 2) H = Bits @ T   (horizontal dilation), this wave computes rows 16*mt..+15
  for (int nt = 0; nt < 4; ++nt) {
    v8f acc = {};
    for (int kc = 0; kc < 2; ++kc) {        // K = 64 in two chunks of 32
      v16h a, b;
      for (int i = 0; i < 16; ++i) {
        int ka = kc * 32 + a_kmap(i, grp);
        a[i] = Bits[(mt * 16 + ln16) * HWDIM + ka];
        int kb = kc * 32 + b_kmap(i, grp);
        b[i] = tval(kb, nt * 16 + ln16);
      }
      acc = __builtin_amdgcn_wmma_f32_16x16x32_f16(false, a, false, b,
                                                   (short)0, acc, false, false);
    }
    // D layout: VGPR r -> row (r + 8*grp), column = lane%16
    for (int r = 0; r < 8; ++r) {
      int y = mt * 16 + r + 8 * grp;
      int x = nt * 16 + ln16;
      Hm[y * HWDIM + x] = (_Float16)acc[r];   // counts <= 7, exact in f16
    }
  }
  __syncthreads();

  // 3) M = T^T @ H    (vertical dilation) -> keep-byte (dropped iff count>=1)
  for (int nt = 0; nt < 4; ++nt) {
    v8f acc = {};
    for (int kc = 0; kc < 2; ++kc) {
      v16h a, b;
      for (int i = 0; i < 16; ++i) {
        int ka = kc * 32 + a_kmap(i, grp);
        a[i] = tval(ka, mt * 16 + ln16);      // T^T[m][k] = T[k][m]
        int kb = kc * 32 + b_kmap(i, grp);
        b[i] = Hm[kb * HWDIM + nt * 16 + ln16];
      }
      acc = __builtin_amdgcn_wmma_f32_16x16x32_f16(false, a, false, b,
                                                   (short)0, acc, false, false);
    }
    for (int r = 0; r < 8; ++r) {
      int y = mt * 16 + r + 8 * grp;
      int x = nt * 16 + ln16;
      maskb[(size_t)c * (HWDIM * HWDIM) + y * HWDIM + x] =
          (acc[r] > 0.5f) ? (unsigned char)0 : (unsigned char)1;
    }
  }
}

// Streaming multiply: 268 MB HBM traffic -> ~11.5 us floor at 23.3 TB/s.
// x/out are touched exactly once -> non-temporal; mask (1 MB) stays in L2.
__global__ void __launch_bounds__(256)
apply_kernel(const float* __restrict__ x, const unsigned char* __restrict__ maskb,
             float* __restrict__ out, long long n4) {
  long long f = (long long)blockIdx.x * blockDim.x + threadIdx.x;
  if (f >= n4) return;
  long long e  = f << 2;                           // first float of this thread
  long long me = e & (long long)(MASK_ELEMS - 1);  // e % (C*H*W), 2^20

  // read-ahead; OOB speculative prefetches are dropped silently (ISA 10.5)
  __builtin_prefetch(x + e + 4096, 0, 0);          // -> global_prefetch_b8

  v4f xv = __builtin_nontemporal_load((const v4f*)(x + e));   // B128, TH=NT
  unsigned int mw = *(const unsigned int*)(maskb + me);        // 4 mask bytes

  v4f r;
  r[0] = xv[0] * (float)( mw        & 0xffu);
  r[1] = xv[1] * (float)((mw >> 8)  & 0xffu);
  r[2] = xv[2] * (float)((mw >> 16) & 0xffu);
  r[3] = xv[3] * (float)((mw >> 24) & 0xffu);
  __builtin_nontemporal_store(r, (v4f*)(out + e));             // B128, TH=NT
}

extern "C" void kernel_launch(void* const* d_in, const int* in_sizes, int n_in,
                              void* d_out, int out_size, void* d_ws, size_t ws_size,
                              hipStream_t stream) {
  const float* x     = (const float*)d_in[0];   // [32,256,64,64] f32
  const float* noise = (const float*)d_in[1];   // [256,58,58]    f32
  float* out         = (float*)d_out;           // [32,256,64,64] f32
  unsigned char* maskb = (unsigned char*)d_ws;  // 1 MB keep-bytes

  (void)in_sizes; (void)n_in; (void)ws_size;

  // Phase 1: mask via WMMA dilation (256 channels x 4 waves, trivial runtime)
  mask_kernel<<<dim3(N_CH), dim3(128), 0, stream>>>(noise, maskb);

  // Phase 2: bandwidth-bound apply (8.39M float4 threads)
  long long n4 = (long long)out_size >> 2;           // 8,388,608
  int blocks = (int)((n4 + 255) / 256);              // 32,768
  apply_kernel<<<dim3(blocks), dim3(256), 0, stream>>>(x, maskb, out, n4);
}
